// LinearMotionDecomposition_12111807774857
// MI455X (gfx1250) — compile-verified, matrix operand order checked by reading, not run on hardware
//
#include <hip/hip_runtime.h>
#include <stdint.h>

#define DIM 512
#define BATCH 32768
#define PB 64                 // QR panel width
#define NPANEL (DIM / PB)

typedef __attribute__((ext_vector_type(16))) __bf16 v16bf;
typedef __attribute__((ext_vector_type(8)))  float  v8f;
typedef __attribute__((ext_vector_type(4)))  unsigned int u32x4;

// ---------- bf16 split helpers (round-to-nearest-even) ----------
__device__ __forceinline__ unsigned short f32_to_bf16_rne(float f) {
  unsigned u = __builtin_bit_cast(unsigned, f);
  u += 0x7FFFu + ((u >> 16) & 1u);
  return (unsigned short)(u >> 16);
}

__device__ __forceinline__ void split_bf16x2(float f, unsigned short& hi, unsigned short& lo) {
  unsigned short h = f32_to_bf16_rne(f);
  float fh = __builtin_bit_cast(float, ((unsigned)h) << 16);
  hi = h;
  lo = f32_to_bf16_rne(f - fh);
}

// Load a 16x32 bf16 WMMA fragment slice for this lane from LDS.
// Per ISA 7.12.2: lane<16 holds K {khalf..khalf+7} and {khalf+16..khalf+23}.
__device__ __forceinline__ v16bf ld_frag(const unsigned short* p) {
  union { u32x4 q[2]; v16bf v; } u;
  u.q[0] = *(const u32x4*)(p);        // 8 bf16 = 16B
  u.q[1] = *(const u32x4*)(p + 16);   // next K-group (+16 elements)
  return u.v;
}

// ---------------------------------------------------------------
// QR kernels: blocked classical Gram-Schmidt (panel width 64)
// ---------------------------------------------------------------
__global__ void copy_f32(const float* __restrict__ src, float* __restrict__ dst, int n) {
  int i = blockIdx.x * blockDim.x + threadIdx.x;
  if (i < n) dst[i] = src[i];
}

// C[i][jj] = sum_k Q[k][i] * Q[k][J+jj]   for i < J, jj < 64
__global__ void qr_proj(const float* __restrict__ Q, float* __restrict__ C, int J) {
  int jj = threadIdx.x;                                  // 0..63
  int i  = blockIdx.y * blockDim.y + threadIdx.y;        // 0..J-1
  float s = 0.f;
  for (int k = 0; k < DIM; ++k)
    s += Q[(size_t)k * DIM + i] * Q[(size_t)k * DIM + J + jj];
  C[i * PB + jj] = s;
}

// Q[k][J+jj] -= sum_{i<J} Q[k][i] * C[i][jj]
__global__ void qr_update(float* __restrict__ Q, const float* __restrict__ C, int J) {
  int jj = threadIdx.x;
  int k  = blockIdx.y * blockDim.y + threadIdx.y;        // 0..511
  float v = Q[(size_t)k * DIM + J + jj];
  for (int i = 0; i < J; ++i)
    v -= Q[(size_t)k * DIM + i] * C[i * PB + jj];
  Q[(size_t)k * DIM + J + jj] = v;
}

// Orthonormalize one 512x64 panel fully inside LDS. One workgroup, 512 threads
// (thread t owns row t). Also applies the sign convention: diag(R)>0 is
// automatic (norms are positive); _adjust_sign == flip column jj by
// sign(Q[J+jj][J+jj]) applied at normalization time.
__global__ __launch_bounds__(DIM) void qr_panel_mgs(float* __restrict__ Q, int J) {
  extern __shared__ float lds[];
  float* P    = lds;                  // [512][64], row stride 65 (bank spread)
  float* cbuf = lds + DIM * 65;       // 64 projection coefficients
  float* part = cbuf + PB;            // 512 partial dots (8 per column)
  float* red  = part + DIM;           // 512-wide reduction buffer
  const int t = threadIdx.x;

  for (int idx = t; idx < DIM * PB; idx += DIM) {
    int k = idx >> 6, j = idx & (PB - 1);
    P[k * 65 + j] = Q[(size_t)k * DIM + J + j];
  }
  __syncthreads();

  for (int jj = 0; jj < PB; ++jj) {
    // dots of column jj against columns 0..jj-1: 8 threads per column
    int g = t >> 3, m = t & 7;
    if (g < jj) {
      float s = 0.f;
      int k0 = m * 64;
      for (int k = k0; k < k0 + 64; ++k) s += P[k * 65 + g] * P[k * 65 + jj];
      part[g * 8 + m] = s;
    }
    __syncthreads();
    if (t < jj) {
      float s = 0.f;
      for (int m2 = 0; m2 < 8; ++m2) s += part[t * 8 + m2];
      cbuf[t] = s;
    }
    __syncthreads();
    // subtract projections from this row's element of column jj
    float v = P[t * 65 + jj];
    for (int i = 0; i < jj; ++i) v -= P[t * 65 + i] * cbuf[i];
    red[t] = v * v;
    __syncthreads();
    for (int s = DIM / 2; s > 0; s >>= 1) {
      if (t < s) red[t] += red[t + s];
      __syncthreads();
    }
    float inv = rsqrtf(red[0]);
    v *= inv;
    __syncthreads();
    if (t == J + jj) red[0] = v;      // broadcast diagonal element
    __syncthreads();
    float sgn = (red[0] < 0.f) ? -1.f : 1.f;
    P[t * 65 + jj] = v * sgn;
    __syncthreads();
  }

  for (int idx = t; idx < DIM * PB; idx += DIM) {
    int k = idx >> 6, j = idx & (PB - 1);
    Q[(size_t)k * DIM + J + j] = P[k * 65 + j];
  }
}

// ---------------------------------------------------------------
// Fused GEMM + add:  out = x + M @ Q^T   (bf16x3 split, f32 accum)
// ---------------------------------------------------------------
#define BM 128
#define BN 128
#define BK 32
#define LDT 40   // LDS ushort stride: multiple of 8 (16B align), bank-spreading

__global__ __launch_bounds__(256) void gemm_bf16x3_add(
    const float* __restrict__ Mg, const float* __restrict__ Qg,
    const float* __restrict__ Xg, float* __restrict__ Og) {
  __shared__ __align__(16) unsigned short sAh[BM * LDT];
  __shared__ __align__(16) unsigned short sAl[BM * LDT];
  __shared__ __align__(16) unsigned short sBh[BN * LDT];
  __shared__ __align__(16) unsigned short sBl[BN * LDT];

  const int t = threadIdx.x;
  const int lane = t & 31, wave = t >> 5;    // 8 waves (wave32)
  const int wRow = wave & 3, wCol = wave >> 2;   // 4x2 wave grid -> 32x64 per wave
  const int half = lane >> 4, l16 = lane & 15;
  const int khalf = half * 8;                // K-group start per ISA A/B layout
  const int rowBase = blockIdx.y * BM;
  const int colBase = blockIdx.x * BN;

  v8f acc[2][4];
#pragma unroll
  for (int mt = 0; mt < 2; ++mt)
#pragma unroll
    for (int nt = 0; nt < 4; ++nt) acc[mt][nt] = (v8f)(0.0f);

  for (int kb = 0; kb < DIM; kb += BK) {
    // Stage 128x32 f32 tiles of M and Q into LDS as hi/lo bf16 (coalesced).
#pragma unroll
    for (int r = 0; r < 16; ++r) {
      int idx = r * 256 + t;
      int row = idx >> 5, k = idx & 31;
      unsigned short h, l;
      split_bf16x2(Mg[(size_t)(rowBase + row) * DIM + kb + k], h, l);
      sAh[row * LDT + k] = h; sAl[row * LDT + k] = l;
      // B = Q^T, so B's N index walks Q's rows: straight row-major copy.
      split_bf16x2(Qg[(size_t)(colBase + row) * DIM + kb + k], h, l);
      sBh[row * LDT + k] = h; sBl[row * LDT + k] = l;
    }
    __syncthreads();

    v16bf ah[2], al[2], bh[4], bl[4];
#pragma unroll
    for (int mt = 0; mt < 2; ++mt) {
      int row = wRow * 32 + mt * 16 + l16;
      ah[mt] = ld_frag(&sAh[row * LDT + khalf]);
      al[mt] = ld_frag(&sAl[row * LDT + khalf]);
    }
#pragma unroll
    for (int nt = 0; nt < 4; ++nt) {
      int col = wCol * 64 + nt * 16 + l16;
      bh[nt] = ld_frag(&sBh[col * LDT + khalf]);
      bl[nt] = ld_frag(&sBl[col * LDT + khalf]);
    }

#pragma unroll
    for (int mt = 0; mt < 2; ++mt)
#pragma unroll
      for (int nt = 0; nt < 4; ++nt) {
        v8f c = acc[mt][nt];
        c = __builtin_amdgcn_wmma_f32_16x16x32_bf16(false, ah[mt], false, bh[nt], (short)0, c, false, false);
        c = __builtin_amdgcn_wmma_f32_16x16x32_bf16(false, ah[mt], false, bl[nt], (short)0, c, false, false);
        c = __builtin_amdgcn_wmma_f32_16x16x32_bf16(false, al[mt], false, bh[nt], (short)0, c, false, false);
        acc[mt][nt] = c;
      }
    __syncthreads();
  }

  // Epilogue: C-tile VGPR r holds rows r (lanes 0-15) / r+8 (lanes 16-31).
#pragma unroll
  for (int mt = 0; mt < 2; ++mt)
#pragma unroll
    for (int nt = 0; nt < 4; ++nt) {
      int row0 = rowBase + wRow * 32 + mt * 16 + half * 8;
      int col  = colBase + wCol * 64 + nt * 16 + l16;
      v8f c = acc[mt][nt];
#pragma unroll
      for (int r = 0; r < 8; ++r) {
        size_t idx = (size_t)(row0 + r) * DIM + col;
        Og[idx] = Xg[idx] + c[r];
      }
    }
}

// ---------------------------------------------------------------
extern "C" void kernel_launch(void* const* d_in, const int* in_sizes, int n_in,
                              void* d_out, int out_size, void* d_ws, size_t ws_size,
                              hipStream_t stream) {
  (void)in_sizes; (void)n_in; (void)out_size; (void)ws_size;
  const float* x   = (const float*)d_in[0];
  const float* mag = (const float*)d_in[1];
  const float* A   = (const float*)d_in[2];
  float* Og = (float*)d_out;
  float* Qw = (float*)d_ws;                  // 512*512 f32 working Q
  float* Cw = Qw + (size_t)DIM * DIM;        // up to 448*64 f32 projections

  copy_f32<<<(DIM * DIM + 255) / 256, 256, 0, stream>>>(A, Qw, DIM * DIM);

  const size_t panel_lds = (size_t)(DIM * 65 + PB + DIM + DIM) * sizeof(float);
  for (int p = 0; p < NPANEL; ++p) {
    int J = p * PB;
    if (J > 0) {
      dim3 blk(PB, 4);
      qr_proj<<<dim3(1, J / 4), blk, 0, stream>>>(Qw, Cw, J);
      qr_update<<<dim3(1, DIM / 4), blk, 0, stream>>>(Qw, Cw, J);
    }
    qr_panel_mgs<<<1, DIM, panel_lds, stream>>>(Qw, J);
  }

  gemm_bf16x3_add<<<dim3(DIM / BN, BATCH / BM), 256, 0, stream>>>(mag, Qw, x, Og);
}
